// StateSpaceLayer_88880053223660
// MI455X (gfx1250) — matrix operand order repaired and checked by reading, not compiled, not run on hardware
//
#include <hip/hip_runtime.h>

#define BB 2
#define SS 512
#define DD 64

typedef __attribute__((ext_vector_type(16))) _Float16 v16h;
typedef __attribute__((ext_vector_type(8)))  float    v8f;

__device__ __forceinline__ float softplus_f(float x) {
    // matches jax.nn.softplus == logaddexp(x, 0)
    return fmaxf(x, 0.0f) + log1pf(__expf(-fabsf(x)));
}

// ---------------------------------------------------------------------------
// Phase 1: per-(b,s) LayerNorm over 64x64, emit xn (f16, natural layout) and
//          log_A_dt transposed to [b][d][s] for the cumsum pass.
// ---------------------------------------------------------------------------
__global__ __launch_bounds__(256) void ssm_phase1(
    const float* __restrict__ x, const float* __restrict__ log_A,
    const float* __restrict__ dt_w, const float* __restrict__ dt_b,
    const float* __restrict__ ln_w, const float* __restrict__ ln_b,
    _Float16* __restrict__ xn_h, float* __restrict__ ladtT)
{
    const int bs = blockIdx.x;          // b*SS + s
    const int b  = bs / SS, s = bs % SS;
    const int t  = threadIdx.x;
    const int i0 = t * 16;              // 16 contiguous elements, within one row d

    const float* xp = x + (size_t)bs * 4096 + i0;
    float v[16];
    float sum = 0.f, sq = 0.f;
    #pragma unroll
    for (int k = 0; k < 4; ++k) {
        float4 f = ((const float4*)xp)[k];
        v[4*k+0] = f.x; v[4*k+1] = f.y; v[4*k+2] = f.z; v[4*k+3] = f.w;
    }
    #pragma unroll
    for (int k = 0; k < 16; ++k) { sum += v[k]; sq += v[k] * v[k]; }

    __shared__ float2 red[256];
    red[t] = make_float2(sum, sq);
    __syncthreads();
    for (int off = 128; off > 0; off >>= 1) {
        if (t < off) {
            red[t].x += red[t + off].x;
            red[t].y += red[t + off].y;
        }
        __syncthreads();
    }
    const float mean = red[0].x * (1.0f / 4096.0f);
    const float var  = red[0].y * (1.0f / 4096.0f) - mean * mean;
    const float rs   = rsqrtf(var + 1e-5f);

    float dot = 0.f;
    _Float16* op = xn_h + (size_t)bs * 4096 + i0;
    #pragma unroll
    for (int k = 0; k < 16; ++k) {
        const int i = i0 + k;
        const float xn = (v[k] - mean) * rs * ln_w[i] + ln_b[i];
        op[k] = (_Float16)xn;
        dot += xn * dt_w[i & 63];
    }
    // 4 consecutive lanes cover one row d -> shuffle reduce within quad
    dot += __shfl_xor(dot, 1, 32);
    dot += __shfl_xor(dot, 2, 32);
    if ((t & 3) == 0) {
        const int d = t >> 2;
        const float lin  = dot + dt_b[0];
        const float dtv  = softplus_f(lin);
        const float ladt = -__expf(log_A[d]) * dtv;
        ladtT[((size_t)b * DD + d) * SS + s] = ladt;
    }
}

// ---------------------------------------------------------------------------
// Phase 2: cumsum over s for each (b,d). One wave per series of 512.
// ---------------------------------------------------------------------------
__global__ __launch_bounds__(32) void ssm_cumsum(const float* __restrict__ ladtT,
                                                 float* __restrict__ csT)
{
    const int bd   = blockIdx.x;
    const int lane = threadIdx.x;
    const float* src = ladtT + (size_t)bd * SS + lane * 16;
    float v[16];
    #pragma unroll
    for (int i = 0; i < 16; ++i) v[i] = src[i];
    #pragma unroll
    for (int i = 1; i < 16; ++i) v[i] += v[i - 1];
    const float tot = v[15];
    float scan = tot;
    #pragma unroll
    for (int off = 1; off < 32; off <<= 1) {
        float n = __shfl_up(scan, off, 32);
        if (lane >= off) scan += n;
    }
    const float add = scan - tot;   // exclusive offset for this lane
    float* dst = csT + (size_t)bd * SS + lane * 16;
    #pragma unroll
    for (int i = 0; i < 16; ++i) dst[i] = v[i] + add;
}

// ---------------------------------------------------------------------------
// Phase 2b: transpose xn to xnT[b][r][c][j] (j contiguous) so WMMA B-operand
//           fragments are two contiguous 16B loads per lane.
// ---------------------------------------------------------------------------
__global__ __launch_bounds__(256) void ssm_transpose(const _Float16* __restrict__ xn_h,
                                                     _Float16* __restrict__ xnT)
{
    const int id = blockIdx.x;
    const int br = id >> 3;              // b*64 + r
    const int j0 = (id & 7) * 64;        // 64-row j tile
    const int b  = br >> 6, r = br & 63;
    const int t  = threadIdx.x;

    __shared__ __align__(16) _Float16 ldsT[64][72];  // [c][j], padded rows

    const _Float16* src = xn_h + (size_t)b * SS * 4096 + (size_t)r * 64;
    #pragma unroll
    for (int h = 0; h < 2; ++h) {
        const int jj = (t >> 3) + h * 32;
        const int c0 = (t & 7) * 8;
        uint4 q = *(const uint4*)(src + (size_t)(j0 + jj) * 4096 + c0);
        const _Float16* hp = (const _Float16*)&q;
        #pragma unroll
        for (int i = 0; i < 8; ++i) ldsT[c0 + i][jj] = hp[i];
    }
    __syncthreads();

    const int c  = t >> 2;
    const int js = (t & 3) * 16;
    _Float16* dst = xnT + ((size_t)br * 64 + c) * SS + j0 + js;
    const uint4* sp = (const uint4*)&ldsT[c][js];
    ((uint4*)dst)[0] = sp[0];
    ((uint4*)dst)[1] = sp[1];
}

// ---------------------------------------------------------------------------
// One K-step (32 j-values): issue all 4 B-fragment load pairs first (one big
// clause), build A = exp(cs_t - cs_j) from LDS while loads are in flight,
// then issue 4 WMMAs with partial loadcnt waits. MASK is compile-time so the
// steady-state loop carries no causal-mask VALU work.
// ---------------------------------------------------------------------------
template <bool MASK>
__device__ __forceinline__ void ssm_kstep(
    int j0, float cs_row, int trow, int hi, int lm,
    const float* lds_cs, const _Float16* colsbase, v8f acc[4])
{
    // ---- B fragments (32x16 f16): half i -> K = i + hi*16, col = q*16+lm
    v16h bf[4];
    #pragma unroll
    for (int q = 0; q < 4; ++q) {
        const _Float16* p = colsbase + (size_t)(q * 16 + lm) * SS + j0 + hi * 16;
        ((uint4*)&bf[q])[0] = ((const uint4*)p)[0];
        ((uint4*)&bf[q])[1] = ((const uint4*)p)[1];
    }

    // ---- A fragment (16x32 f16): half i -> K = i + hi*16
    const float4 f0 = *(const float4*)&lds_cs[j0 + hi * 16 + 0];
    const float4 f1 = *(const float4*)&lds_cs[j0 + hi * 16 + 4];
    const float4 f2 = *(const float4*)&lds_cs[j0 + hi * 16 + 8];
    const float4 f3 = *(const float4*)&lds_cs[j0 + hi * 16 + 12];
    const float cj[16] = {f0.x, f0.y, f0.z, f0.w, f1.x, f1.y, f1.z, f1.w,
                          f2.x, f2.y, f2.z, f2.w, f3.x, f3.y, f3.z, f3.w};
    v16h a;
    #pragma unroll
    for (int i = 0; i < 16; ++i) {
        const int K = i + hi * 16;
        float m = __expf(cs_row - cj[i]);       // <= 1 in causal region
        if (MASK) {
            if (j0 + K > trow) m = 0.0f;
        }
        a[i] = (_Float16)m;
    }

    // ---- 4 WMMAs sharing the A fragment
    #pragma unroll
    for (int q = 0; q < 4; ++q)
        acc[q] = __builtin_amdgcn_wmma_f32_16x16x32_f16(
            false, a, false, bf[q], (short)0, acc[q], false, false);
}

// ---------------------------------------------------------------------------
// Phase 3: causal decay-GEMM. One block per (b,r, 128-row t-block); 8 waves,
// each wave owns a 16(t) x 64(c) strip. A = exp(cs_t - cs_j) built on the fly
// from LDS-staged cs (TRANS v_exp co-executes with XDL WMMA); B from xnT.
// ---------------------------------------------------------------------------
__global__ __launch_bounds__(256) void ssm_gemm(
    const float* __restrict__ csT, const _Float16* __restrict__ xnT,
    const float* __restrict__ x, float* __restrict__ out)
{
    const int br = blockIdx.x;           // b*64 + r
    const int b  = br >> 6, r = br & 63;

    __shared__ __align__(16) float lds_cs[SS];
    for (int i = threadIdx.x; i < SS; i += 256)
        lds_cs[i] = csT[(size_t)br * SS + i];
    __syncthreads();

    const int wave = threadIdx.x >> 5;
    const int lane = threadIdx.x & 31;
    const int hi   = lane >> 4;          // lane half (0/1)
    const int lm   = lane & 15;
    const int t0   = blockIdx.y * 128 + wave * 16;
    const int trow = t0 + lm;            // A-matrix row for this lane
    const float cs_row = lds_cs[trow];

    const _Float16* colsbase = xnT + (size_t)br * DD * SS;  // [c][j]

    v8f acc[4] = {};

    // full (mask-free) tiles: j0 + 31 <= t0
    const int jFullEnd = (t0 >= 31) ? ((((t0 - 31) >> 5) + 1) << 5) : 0;
    int j0 = 0;
    for (; j0 < jFullEnd; j0 += 32)
        ssm_kstep<false>(j0, cs_row, trow, hi, lm, lds_cs, colsbase, acc);
    // diagonal (masked) tiles
    for (; j0 <= t0 + 15; j0 += 32)
        ssm_kstep<true>(j0, cs_row, trow, hi, lm, lds_cs, colsbase, acc);

    // ---- epilogue: out = acc + x_in (C layout: vgpr p -> row p + hi*8)
    #pragma unroll
    for (int q = 0; q < 4; ++q) {
        const int c = q * 16 + lm;
        #pragma unroll
        for (int p = 0; p < 8; ++p) {
            const int tt = t0 + hi * 8 + p;
            const size_t idx = (((size_t)b * SS + tt) * DD + r) * DD + c;
            out[idx] = acc[q][p] + x[idx];
        }
    }
}

// ---------------------------------------------------------------------------
extern "C" void kernel_launch(void* const* d_in, const int* in_sizes, int n_in,
                              void* d_out, int out_size, void* d_ws, size_t ws_size,
                              hipStream_t stream) {
    const float* x    = (const float*)d_in[0];
    const float* logA = (const float*)d_in[1];
    const float* dt_w = (const float*)d_in[2];
    const float* dt_b = (const float*)d_in[3];
    const float* ln_w = (const float*)d_in[4];
    const float* ln_b = (const float*)d_in[5];
    float* out = (float*)d_out;

    char* ws = (char*)d_ws;
    _Float16* xn_h = (_Float16*)ws;                                     // 8 MB
    _Float16* xnT  = (_Float16*)(ws + (size_t)(8u << 20));              // 8 MB
    float*    ladt = (float*)(ws + (size_t)(16u << 20));                // 256 KB
    float*    csT  = (float*)(ws + (size_t)(16u << 20) + (256u << 10)); // 256 KB

    ssm_phase1<<<dim3(BB * SS), dim3(256), 0, stream>>>(
        x, logA, dt_w, dt_b, ln_w, ln_b, xn_h, ladt);
    ssm_cumsum<<<dim3(BB * DD), dim3(32), 0, stream>>>(ladt, csT);
    ssm_transpose<<<dim3(BB * DD * 8), dim3(256), 0, stream>>>(xn_h, xnT);
    ssm_gemm<<<dim3(BB * DD, SS / 128), dim3(256), 0, stream>>>(csT, xnT, x, out);
}
